// AttentionSP_50852412784805
// MI455X (gfx1250) — matrix operand order repaired and checked by reading, not compiled, not run on hardware
//
#include <hip/hip_runtime.h>

typedef __bf16 bf16;
typedef __attribute__((ext_vector_type(16))) __bf16 v16bf;
typedef __attribute__((ext_vector_type(8)))  __bf16 v8bf;
typedef __attribute__((ext_vector_type(8)))  float  v8f;
typedef __attribute__((ext_vector_type(4)))  unsigned int v4u;
typedef __attribute__((ext_vector_type(8)))  int v8i;
typedef __attribute__((ext_vector_type(4)))  int v4i;

// Sizes: B=1, N=4096, C=768, H=12, D=64, 3C=2304
#define SEQ 4096
#define CDIM 768
#define NHEAD 12
#define HDIM 64

// ---------------------------------------------------------------------------
// Fragment loaders per CDNA5 ISA 7.12.2 (wave32)
// A-matrix 16x32 bf16: lane L -> row = L%16; halves 0..7 = K kb..kb+7,
// halves 8..15 = K kb+16..kb+23, kb = (L<16 ? 0 : 8).  src row-major [M][K].
static __device__ __forceinline__ v16bf load_a_frag(const bf16* __restrict__ src,
                                                    int ld, int row0, int k0) {
  const int lane = threadIdx.x & 31;
  const int row  = row0 + (lane & 15);
  const int kb   = k0 + ((lane & 16) ? 8 : 0);
  const bf16* p  = src + (size_t)row * ld + kb;
  v8bf lo = *(const v8bf*)(p);
  v8bf hi = *(const v8bf*)(p + 16);
  return __builtin_shufflevector(lo, hi, 0,1,2,3,4,5,6,7,8,9,10,11,12,13,14,15);
}

// B-matrix 32x16 bf16: lane L -> col = L%16; halves 0..7 = K kb..kb+7,
// halves 8..15 = K kb+16..kb+23, kb = (L<16 ? 0 : 8). srcT stored row-major [Ncols][K] (B transposed).
static __device__ __forceinline__ v16bf load_b_frag(const bf16* __restrict__ srcT,
                                                    int ld, int col0, int k0) {
  const int lane = threadIdx.x & 31;
  const int col  = col0 + (lane & 15);
  const int kb   = k0 + ((lane & 16) ? 8 : 0);
  const bf16* p  = srcT + (size_t)col * ld + kb;
  v8bf lo = *(const v8bf*)(p);
  v8bf hi = *(const v8bf*)(p + 16);
  return __builtin_shufflevector(lo, hi, 0,1,2,3,4,5,6,7,8,9,10,11,12,13,14,15);
}

static __device__ __forceinline__ v8f wmma_bf16(v16bf a, v16bf b, v8f c) {
  return __builtin_amdgcn_wmma_f32_16x16x32_bf16(false, a, false, b,
                                                 (short)0, c, false, false);
}

// ---------------------------------------------------------------------------
// Tensor Data Mover: 2D tile load, data_size = 8 bytes (dims in 8B units).
// D# layout per CDNA5 ISA; groups 2/3 zero (2-D).
static __device__ __forceinline__ void tdm_load_2d(unsigned int lds_off,
                                                   const void* gptr,
                                                   unsigned int tile_d0,   // 8B units per row
                                                   unsigned int tile_d1,   // rows
                                                   unsigned int stride0) { // row stride, 8B units
  const unsigned long long ga = (unsigned long long)(uintptr_t)gptr;
  v4u g0;
  g0.x = 1u;                                   // count=1, user descriptor
  g0.y = lds_off;                              // lds_addr (bytes)
  g0.z = (unsigned int)ga;                     // global_addr[31:0]
  g0.w = (unsigned int)(ga >> 32) | (2u << 30);// global_addr[56:32] | type=2
  v8i g1;
  g1[0] = (int)(3u << 16);                     // workgroup_mask=0, data_size=8B
  g1[1] = (int)((tile_d0 & 0xFFFFu) << 16);    // abar=0 | tensor_dim0[15:0]
  g1[2] = (int)((tile_d0 >> 16) | ((tile_d1 & 0xFFFFu) << 16)); // dim0 hi | dim1 lo
  g1[3] = (int)((tile_d1 >> 16) | ((tile_d0 & 0xFFFFu) << 16)); // dim1 hi | tile_dim0
  g1[4] = (int)(tile_d1 & 0xFFFFu);            // tile_dim1 | tile_dim2=0
  g1[5] = (int)stride0;                        // tensor_dim0_stride[31:0]
  g1[6] = 0;                                   // stride0 hi | stride1 lo
  g1[7] = 0;                                   // stride1 hi
  v4i g2 = {0, 0, 0, 0};
  v4i g3 = {0, 0, 0, 0};
#if defined(__clang_major__) && (__clang_major__ >= 23)
  v8i g4 = {0, 0, 0, 0, 0, 0, 0, 0};
  __builtin_amdgcn_tensor_load_to_lds(g0, g1, g2, g3, g4, 0);
#else
  __builtin_amdgcn_tensor_load_to_lds(g0, g1, g2, g3, 0);
#endif
}

// ---------------------------------------------------------------------------
// Conversion kernels
__global__ void cvt_f32_bf16_kernel(const float* __restrict__ in,
                                    bf16* __restrict__ out, int n) {
  int i = blockIdx.x * blockDim.x + threadIdx.x;
  if (i < n) out[i] = (bf16)in[i];
}

// out[c][r] = (bf16) in[r][c]  (in: rows x cols f32 row-major)
__global__ void cvt_transpose_kernel(const float* __restrict__ in,
                                     bf16* __restrict__ out, int rows, int cols) {
  int idx = blockIdx.x * blockDim.x + threadIdx.x;
  if (idx >= rows * cols) return;
  int c = idx / rows;
  int r = idx - c * rows;
  out[idx] = (bf16)in[(size_t)r * cols + c];
}

// ---------------------------------------------------------------------------
// QKV GEMM, 32x32 output per wave (2x2 register blocking -> 2x A/B reuse).
// qkv col c = s*768 + h*64 + d.  Q pre-scaled; V stored transposed [H][D][N].
__global__ void __launch_bounds__(256) qkv_gemm_kernel(const bf16* __restrict__ xb,
                                                       const bf16* __restrict__ wT,
                                                       bf16* __restrict__ Qb,
                                                       bf16* __restrict__ Kb,
                                                       bf16* __restrict__ Vt) {
  const int lane = threadIdx.x & 31;
  const int wid  = blockIdx.x * 8 + (threadIdx.x >> 5);
  const int ct   = wid % 72;           // 2304/32
  const int rt   = wid / 72;           // 4096/32
  const int n0 = rt * 32, c0 = ct * 32;
  v8f acc[2][2];
  const v8f vzero = {};
#pragma unroll
  for (int i = 0; i < 2; ++i)
#pragma unroll
    for (int j = 0; j < 2; ++j) acc[i][j] = vzero;

  for (int k0 = 0; k0 < CDIM; k0 += 32) {
    v16bf a0 = load_a_frag(xb, CDIM, n0, k0);
    v16bf a1 = load_a_frag(xb, CDIM, n0 + 16, k0);
    v16bf b0 = load_b_frag(wT, CDIM, c0, k0);
    v16bf b1 = load_b_frag(wT, CDIM, c0 + 16, k0);
    acc[0][0] = wmma_bf16(a0, b0, acc[0][0]);
    acc[0][1] = wmma_bf16(a0, b1, acc[0][1]);
    acc[1][0] = wmma_bf16(a1, b0, acc[1][0]);
    acc[1][1] = wmma_bf16(a1, b1, acc[1][1]);
  }
  // 32-wide col tile never crosses s (768) or head (64) boundaries.
  const int s   = c0 / CDIM;
  const int rem = c0 - s * CDIM;
  const int hh  = rem >> 6;
  const int d0  = rem & 63;
  const int col = lane & 15;
  const int rbase = (lane & 16) ? 8 : 0;
  if (s == 0) {
#pragma unroll
    for (int ri = 0; ri < 2; ++ri)
#pragma unroll
      for (int ci = 0; ci < 2; ++ci)
#pragma unroll
        for (int r = 0; r < 8; ++r) {
          const int n = n0 + ri * 16 + rbase + r;
          const int d = d0 + ci * 16 + col;
          Qb[((size_t)hh * SEQ + n) * HDIM + d] = (bf16)(acc[ri][ci][r] * 0.125f);
        }
  } else if (s == 1) {
#pragma unroll
    for (int ri = 0; ri < 2; ++ri)
#pragma unroll
      for (int ci = 0; ci < 2; ++ci)
#pragma unroll
        for (int r = 0; r < 8; ++r) {
          const int n = n0 + ri * 16 + rbase + r;
          const int d = d0 + ci * 16 + col;
          Kb[((size_t)hh * SEQ + n) * HDIM + d] = (bf16)acc[ri][ci][r];
        }
  } else {
#pragma unroll
    for (int ri = 0; ri < 2; ++ri)
#pragma unroll
      for (int ci = 0; ci < 2; ++ci)
#pragma unroll
        for (int r = 0; r < 8; ++r) {
          const int n = n0 + ri * 16 + rbase + r;
          const int d = d0 + ci * 16 + col;
          Vt[((size_t)hh * HDIM + d) * SEQ + n] = (bf16)acc[ri][ci][r];
        }
  }
}

// ---------------------------------------------------------------------------
// Flash attention: block = 8 waves = one head x 128 queries.  K/V 32-key tiles
// staged into LDS by the Tensor Data Mover (wave 0 issues; double-buffered;
// s_wait_tensorcnt + workgroup barrier).  K/V global traffic amortized 8x.
__global__ void __launch_bounds__(256) attn_kernel(const bf16* __restrict__ Q,
                                                   const bf16* __restrict__ K,
                                                   const bf16* __restrict__ Vt,
                                                   bf16* __restrict__ Aout) {
  __shared__ __align__(16) bf16 kbuf[2][32 * HDIM];   // [buf][key][d]   8KB
  __shared__ __align__(16) bf16 vbuf[2][HDIM * 32];   // [buf][d][key]   8KB
  __shared__ float pbuf[8][16 * 32];                  // per-wave P      16KB

  const int lane  = threadIdx.x & 31;
  const int wslot = threadIdx.x >> 5;
  const int h  = blockIdx.x >> 5;                     // / (4096/128)
  const int q0 = (blockIdx.x & 31) * 128 + wslot * 16;

  const bf16* Qh = Q  + (size_t)h * SEQ * HDIM;
  const bf16* Kh = K  + (size_t)h * SEQ * HDIM;
  const bf16* Vh = Vt + (size_t)h * HDIM * SEQ;

  const v16bf aq0 = load_a_frag(Qh, HDIM, q0, 0);
  const v16bf aq1 = load_a_frag(Qh, HDIM, q0, 32);

  float m[8], l[8];
  v8f o[4];
  const v8f vzero = {};
#pragma unroll
  for (int r = 0; r < 8; ++r) { m[r] = -3.0e38f; l[r] = 0.0f; }
#pragma unroll
  for (int t = 0; t < 4; ++t) o[t] = vzero;

  float* pw = &pbuf[wslot][0];
  const float* prd = pw + (lane & 15) * 32 + ((lane & 16) ? 8 : 0);
  const int rowbase = (lane & 16) ? 8 : 0;
  const bool issuer = (wslot == 0);                   // wave-uniform branch

  // Prologue: stage key-block 0 into buffer 0.
  if (issuer) {
    tdm_load_2d((unsigned int)(uintptr_t)&kbuf[0][0], Kh, HDIM / 4, 32, HDIM / 4);
    tdm_load_2d((unsigned int)(uintptr_t)&vbuf[0][0], Vh, 8, HDIM, SEQ / 4);
    __builtin_amdgcn_s_wait_tensorcnt(0);
  }
  __syncthreads();

  for (int it = 0; it < SEQ / 32; ++it) {
    const int cur = it & 1;
    if (issuer && (it + 1) < SEQ / 32) {              // prefetch next block
      const int j1 = (it + 1) * 32;
      tdm_load_2d((unsigned int)(uintptr_t)&kbuf[cur ^ 1][0],
                  Kh + (size_t)j1 * HDIM, HDIM / 4, 32, HDIM / 4);
      tdm_load_2d((unsigned int)(uintptr_t)&vbuf[cur ^ 1][0],
                  Vh + j1, 8, HDIM, SEQ / 4);
    }
    const bf16* kt = &kbuf[cur][0];                   // [32][64] row-major
    const bf16* vt = &vbuf[cur][0];                   // [64][32] row-major

    // S = (Q*scale) . K^T for the 32 staged keys.
    v8f s0 = vzero, s1 = vzero;
    s0 = wmma_bf16(aq0, load_b_frag(kt, HDIM, 0, 0),   s0);
    s0 = wmma_bf16(aq1, load_b_frag(kt, HDIM, 0, 32),  s0);
    s1 = wmma_bf16(aq0, load_b_frag(kt, HDIM, 16, 0),  s1);
    s1 = wmma_bf16(aq1, load_b_frag(kt, HDIM, 16, 32), s1);

    // Online softmax (row reductions across 16-lane halves).
#pragma unroll
    for (int r = 0; r < 8; ++r) {
      const float v0 = s0[r], v1 = s1[r];
      float mx = fmaxf(v0, v1);
#pragma unroll
      for (int off = 8; off >= 1; off >>= 1)
        mx = fmaxf(mx, __shfl_xor(mx, off, 32));
      const float mnew  = fmaxf(m[r], mx);
      const float alpha = __expf(m[r] - mnew);
      const float p0 = __expf(v0 - mnew);
      const float p1 = __expf(v1 - mnew);
      float rs = p0 + p1;
#pragma unroll
      for (int off = 8; off >= 1; off >>= 1)
        rs += __shfl_xor(rs, off, 32);
      l[r] = l[r] * alpha + rs;
      m[r] = mnew;
#pragma unroll
      for (int t = 0; t < 4; ++t) o[t][r] *= alpha;
      const int row = rowbase + r;
      pw[row * 32 + (lane & 15)]      = p0;
      pw[row * 32 + 16 + (lane & 15)] = p1;
    }
    // Re-read P in A-fragment order (wave-private LDS region; per-wave DS ops
    // are in-order; compiler barrier prevents reordering).
    asm volatile("" ::: "memory");
    v16bf ap;
#pragma unroll
    for (int i = 0; i < 8; ++i) {
      ap[i]     = (bf16)prd[i];
      ap[8 + i] = (bf16)prd[16 + i];
    }
    asm volatile("" ::: "memory");
    // O += P . V  (Vt rows in LDS are B columns)
#pragma unroll
    for (int t = 0; t < 4; ++t)
      o[t] = wmma_bf16(ap, load_b_frag(vt, 32, t * 16, 0), o[t]);

    if (issuer) __builtin_amdgcn_s_wait_tensorcnt(0); // next block landed
    __syncthreads();                                  // release all waves
  }

#pragma unroll
  for (int r = 0; r < 8; ++r) {
    const float inv = 1.0f / l[r];
    const int n = q0 + rowbase + r;
    bf16* dst = Aout + (size_t)n * CDIM + h * HDIM + (lane & 15);
#pragma unroll
    for (int t = 0; t < 4; ++t)
      dst[t * 16] = (bf16)(o[t][r] * inv);
  }
}

// ---------------------------------------------------------------------------
// Projection, 32x32 per wave: out = attn_out . w_proj + b_proj (f32 output)
__global__ void __launch_bounds__(256) proj_gemm_kernel(const bf16* __restrict__ A,
                                                        const bf16* __restrict__ wT,
                                                        const float* __restrict__ bias,
                                                        float* __restrict__ out) {
  const int lane = threadIdx.x & 31;
  const int wid  = blockIdx.x * 8 + (threadIdx.x >> 5);
  const int ct   = wid % 24;           // 768/32
  const int rt   = wid / 24;
  const int n0 = rt * 32, c0 = ct * 32;
  v8f acc[2][2];
  const v8f vzero = {};
#pragma unroll
  for (int i = 0; i < 2; ++i)
#pragma unroll
    for (int j = 0; j < 2; ++j) acc[i][j] = vzero;

  for (int k0 = 0; k0 < CDIM; k0 += 32) {
    v16bf a0 = load_a_frag(A, CDIM, n0, k0);
    v16bf a1 = load_a_frag(A, CDIM, n0 + 16, k0);
    v16bf b0 = load_b_frag(wT, CDIM, c0, k0);
    v16bf b1 = load_b_frag(wT, CDIM, c0 + 16, k0);
    acc[0][0] = wmma_bf16(a0, b0, acc[0][0]);
    acc[0][1] = wmma_bf16(a0, b1, acc[0][1]);
    acc[1][0] = wmma_bf16(a1, b0, acc[1][0]);
    acc[1][1] = wmma_bf16(a1, b1, acc[1][1]);
  }
  const int rbase = (lane & 16) ? 8 : 0;
#pragma unroll
  for (int ci = 0; ci < 2; ++ci) {
    const int col = c0 + ci * 16 + (lane & 15);
    const float b = bias[col];
#pragma unroll
    for (int ri = 0; ri < 2; ++ri)
#pragma unroll
      for (int r = 0; r < 8; ++r)
        out[(size_t)(n0 + ri * 16 + rbase + r) * CDIM + col] = acc[ri][ci][r] + b;
  }
}

// ---------------------------------------------------------------------------
extern "C" void kernel_launch(void* const* d_in, const int* in_sizes, int n_in,
                              void* d_out, int out_size, void* d_ws, size_t ws_size,
                              hipStream_t stream) {
  const float* x      = (const float*)d_in[0];  // [4096,768]
  const float* w_qkv  = (const float*)d_in[1];  // [768,2304]
  const float* w_proj = (const float*)d_in[2];  // [768,768]
  const float* b_proj = (const float*)d_in[3];  // [768]
  float* out = (float*)d_out;

  uint8_t* ws = (uint8_t*)d_ws;
  const size_t SZ_XB    = (size_t)SEQ * CDIM * 2;          // 6291456
  const size_t SZ_WQKVT = (size_t)3 * CDIM * CDIM * 2;     // 3538944
  const size_t SZ_WPRJT = (size_t)CDIM * CDIM * 2;         // 1179648
  const size_t SZ_HEADS = (size_t)NHEAD * SEQ * HDIM * 2;  // 6291456

  bf16* xb     = (bf16*)(ws);
  bf16* wqkvT  = (bf16*)(ws + SZ_XB);
  bf16* wprojT = (bf16*)(ws + SZ_XB + SZ_WQKVT);
  bf16* Qb     = (bf16*)(ws + SZ_XB + SZ_WQKVT + SZ_WPRJT);
  bf16* Kb     = (bf16*)(ws + SZ_XB + SZ_WQKVT + SZ_WPRJT + SZ_HEADS);
  bf16* Vt     = (bf16*)(ws + SZ_XB + SZ_WQKVT + SZ_WPRJT + 2 * SZ_HEADS);
  bf16* AOut   = (bf16*)(ws + SZ_XB + SZ_WQKVT + SZ_WPRJT + 3 * SZ_HEADS);

  {
    int n = SEQ * CDIM;
    cvt_f32_bf16_kernel<<<(n + 255) / 256, 256, 0, stream>>>(x, xb, n);
  }
  {
    int n = CDIM * 3 * CDIM;
    cvt_transpose_kernel<<<(n + 255) / 256, 256, 0, stream>>>(w_qkv, wqkvT, CDIM, 3 * CDIM);
  }
  {
    int n = CDIM * CDIM;
    cvt_transpose_kernel<<<(n + 255) / 256, 256, 0, stream>>>(w_proj, wprojT, CDIM, CDIM);
  }
  // QKV GEMM: (4096/32)x(2304/32) = 9216 waves, 8 per block
  qkv_gemm_kernel<<<9216 / 8, 256, 0, stream>>>(xb, wqkvT, Qb, Kb, Vt);
  // Attention: 12 heads x 32 query-blocks = 384 blocks of 8 waves
  attn_kernel<<<NHEAD * (SEQ / 128), 256, 0, stream>>>(Qb, Kb, Vt, AOut);
  // Projection: (4096/32)x(768/32) = 3072 waves
  proj_gemm_kernel<<<3072 / 8, 256, 0, stream>>>(AOut, wprojT, b_proj, out);
}
